// NeuronOptimizedAttention_77841987272939
// MI455X (gfx1250) — matrix operand order repaired and checked by reading, not compiled, not run on hardware
//
#include <hip/hip_runtime.h>

#define D_MODEL 1024
#define NHEAD   16
#define DHEAD   64
#define BATCH   4
#define SEQ     2048
#define MROWS   (BATCH * SEQ)          // 8192
#define SCALE_Q 0.125f                 // 1/sqrt(64)

typedef __bf16 bf16;
typedef __attribute__((ext_vector_type(16))) __bf16 v16bf;
typedef __attribute__((ext_vector_type(8)))  __bf16 v8bf;
typedef __attribute__((ext_vector_type(8)))  float  v8f;
typedef __attribute__((ext_vector_type(4)))  int    v4i;

#if defined(__gfx1250__) && __has_builtin(__builtin_amdgcn_global_load_async_to_lds_b128) && \
    __has_builtin(__builtin_amdgcn_s_wait_asynccnt)
#define USE_ASYNC_LDS 1
#define AS1_V4I(p) ((__attribute__((address_space(1))) v4i*)(p))
#define AS3_V4I(p) ((__attribute__((address_space(3))) v4i*)(p))
#else
#define USE_ASYNC_LDS 0
#endif

// ---------------------------------------------------------------------------
// Fragment loaders (wave32 WMMA 16x16x32 bf16 layouts, ISA 7.12.2)
// ---------------------------------------------------------------------------

// A-matrix 16x32 (MxK), src row-major [row][k], leading dim ld.
// lane&15 = row; lanes 0-15 hold K {0..7,16..23}, lanes 16-31 hold K {8..15,24..31}.
__device__ __forceinline__ v16bf load_a_frag(const bf16* src, int ld,
                                             int row0, int k0, int lane) {
    const int r  = lane & 15;
    const int hf = lane >> 4;
    const bf16* p = src + (size_t)(row0 + r) * ld + k0 + hf * 8;
    v8bf c0 = *(const v8bf*)p;          // K = k0 + hf*8 + 0..7
    v8bf c1 = *(const v8bf*)(p + 16);   // K = k0 + 16 + hf*8 + 0..7
    v16bf out;
#pragma unroll
    for (int i = 0; i < 8; ++i) { out[i] = c0[i]; out[8 + i] = c1[i]; }
    return out;
}

// B-matrix 32x16 (KxN) where B[k][n] = src[n][k] (src row-major NxK).
// lane&15 = n; lanes 0-15 hold K {0..7,16..23}, lanes 16-31 hold K {8..15,24..31}.
__device__ __forceinline__ v16bf load_bT_frag(const bf16* src, int ld,
                                              int n0, int k0, int lane) {
    const int n  = n0 + (lane & 15);
    const int hf = lane >> 4;
    const bf16* p = src + (size_t)n * ld + k0 + hf * 8;
    v8bf c0 = *(const v8bf*)p;
    v8bf c1 = *(const v8bf*)(p + 16);
    v16bf out;
#pragma unroll
    for (int i = 0; i < 8; ++i) { out[i] = c0[i]; out[8 + i] = c1[i]; }
    return out;
}

// ---------------------------------------------------------------------------
// fp32 -> bf16 conversion
// ---------------------------------------------------------------------------
__global__ void cvt_f32_bf16_kernel(const float* __restrict__ in,
                                    bf16* __restrict__ out, int n) {
    int i = (blockIdx.x * blockDim.x + threadIdx.x) * 4;
    if (i < n) {
        float4 f = *(const float4*)(in + i);
        out[i + 0] = (bf16)f.x; out[i + 1] = (bf16)f.y;
        out[i + 2] = (bf16)f.z; out[i + 3] = (bf16)f.w;
    }
}

// ---------------------------------------------------------------------------
// GEMM:  out = (A[M,K] x W[N,K]^T + bias) * scale
// mode 0: bf16 into [B, H, S, Dh]          (Q, K projections)
// mode 1: f32 row-major [M, N]             (final O projection)
// mode 2: bf16 transposed [B, H, Dh, S]    (V projection -> contiguous PV frags)
// Block = 256 threads (8 waves); block tile 256(M) x 128(N); wave tile 64x64.
// ---------------------------------------------------------------------------
__global__ void __launch_bounds__(256)
gemm_bf16_kernel(const bf16* __restrict__ A, const bf16* __restrict__ W,
                 const float* __restrict__ bias, bf16* __restrict__ outb,
                 float* __restrict__ outf, int mode, float scale) {
    const int lane = threadIdx.x & 31;
    const int wave = threadIdx.x >> 5;
    const int m0 = blockIdx.x * 256 + (wave >> 1) * 64;
    const int n0 = blockIdx.y * 128 + (wave & 1) * 64;

    v8f acc[4][4] = {};
    for (int kb = 0; kb < D_MODEL; kb += 32) {
        v16bf a[4], b[4];
#pragma unroll
        for (int i = 0; i < 4; ++i) a[i] = load_a_frag(A, D_MODEL, m0 + 16 * i, kb, lane);
#pragma unroll
        for (int j = 0; j < 4; ++j) b[j] = load_bT_frag(W, D_MODEL, n0 + 16 * j, kb, lane);
#pragma unroll
        for (int i = 0; i < 4; ++i)
#pragma unroll
            for (int j = 0; j < 4; ++j)
                acc[i][j] = __builtin_amdgcn_wmma_f32_16x16x32_bf16(
                    false, a[i], false, b[j], (short)0, acc[i][j], false, false);
    }

    const int cn = lane & 15, hf = lane >> 4;
#pragma unroll
    for (int i = 0; i < 4; ++i)
#pragma unroll
        for (int j = 0; j < 4; ++j) {
            const int col = n0 + 16 * j + cn;
            const float bvv = bias[col];
#pragma unroll
            for (int r = 0; r < 8; ++r) {
                const int row = m0 + 16 * i + r + 8 * hf;
                const float v = (acc[i][j][r] + bvv) * scale;
                const int b_ = row >> 11, s = row & (SEQ - 1);
                const int hd = col >> 6,  d = col & (DHEAD - 1);
                if (mode == 0) {
                    outb[(((size_t)(b_ * NHEAD + hd)) * SEQ + s) * DHEAD + d] = (bf16)v;
                } else if (mode == 2) {
                    outb[(((size_t)(b_ * NHEAD + hd)) * DHEAD + d) * SEQ + s] = (bf16)v;
                } else {
                    outf[(size_t)row * D_MODEL + col] = v;
                }
            }
        }
}

// ---------------------------------------------------------------------------
// Flash attention over [B*H] heads; Q pre-scaled by 1/sqrt(Dh).
// Block = 8 waves sharing one head; each wave owns 16 Q rows.
// KV blocks of 32 staged into LDS (async global->LDS when available).
// Output bf16 [B, S, H*Dh].
// ---------------------------------------------------------------------------
__global__ void __launch_bounds__(256)
attn_kernel(const bf16* __restrict__ Q, const bf16* __restrict__ K,
            const bf16* __restrict__ Vt, bf16* __restrict__ O) {
    __shared__ bf16 ldsK[32 * 72];       // K tile  [t_local][d],  stride 72
    __shared__ bf16 ldsV[64 * 40];       // V^T tile [d][t_local], stride 40
    __shared__ bf16 ldsP[8 * 16 * 40];   // per-wave P bounce (16x32), stride 40

    const int tid  = threadIdx.x;
    const int lane = tid & 31;
    const int wave = tid >> 5;
    const int bh = blockIdx.x;                       // 0..63
    const int q0 = (blockIdx.y * 8 + wave) * 16;
    const bf16* Qb = Q  + (size_t)bh * SEQ * DHEAD;
    const bf16* Kb = K  + (size_t)bh * SEQ * DHEAD;
    const bf16* Vb = Vt + (size_t)bh * DHEAD * SEQ;  // [d][t]
    bf16* ldsW = ldsP + wave * 16 * 40;
    const int cn = lane & 15, hf = lane >> 4;

    // cooperative-staging coordinates (256 threads, 16B chunks)
    const int krow = tid >> 3, kcs = (tid & 7) * 8;  // K tile: 32 rows x 64
    const int vrow = tid >> 2, vcs = (tid & 3) * 8;  // V tile: 64 rows x 32

    const v16bf qa0 = load_a_frag(Qb, DHEAD, q0, 0,  lane);
    const v16bf qa1 = load_a_frag(Qb, DHEAD, q0, 32, lane);

    v8f acc[4] = {};
    float mrow[8], lrow[8];
#pragma unroll
    for (int r = 0; r < 8; ++r) { mrow[r] = -1e30f; lrow[r] = 0.0f; }

    for (int t0 = 0; t0 < SEQ; t0 += 32) {
        // ---- stage K (32x64) and V^T (64x32) tiles into LDS ----
#if USE_ASYNC_LDS
        __builtin_amdgcn_global_load_async_to_lds_b128(
            AS1_V4I(Kb + (size_t)(t0 + krow) * DHEAD + kcs),
            AS3_V4I(&ldsK[krow * 72 + kcs]), 0, 0);
        __builtin_amdgcn_global_load_async_to_lds_b128(
            AS1_V4I(Vb + (size_t)vrow * SEQ + t0 + vcs),
            AS3_V4I(&ldsV[vrow * 40 + vcs]), 0, 0);
#else
        {
            v8bf kk = *(const v8bf*)(Kb + (size_t)(t0 + krow) * DHEAD + kcs);
            v8bf vv = *(const v8bf*)(Vb + (size_t)vrow * SEQ + t0 + vcs);
            *(v8bf*)(&ldsK[krow * 72 + kcs]) = kk;
            *(v8bf*)(&ldsV[vrow * 40 + vcs]) = vv;
        }
#endif
        if (t0 + 32 < SEQ) {  // prefetch next KV tiles (global_prefetch_b8)
            __builtin_prefetch(Kb + (size_t)(t0 + 32 + krow) * DHEAD + kcs, 0, 0);
            __builtin_prefetch(Vb + (size_t)vrow * SEQ + t0 + 32 + vcs, 0, 0);
        }
#if USE_ASYNC_LDS
        __builtin_amdgcn_s_wait_asynccnt(0);  // own async copies landed in LDS
#endif
        __syncthreads();

        // ---- S = Q x K^T  (16 x 32 scores) ----
        v8f sc[2] = {};
#pragma unroll
        for (int tb = 0; tb < 2; ++tb) {
            v16bf kf0 = load_bT_frag(ldsK, 72, 16 * tb, 0,  lane);
            v16bf kf1 = load_bT_frag(ldsK, 72, 16 * tb, 32, lane);
            sc[tb] = __builtin_amdgcn_wmma_f32_16x16x32_bf16(
                false, qa0, false, kf0, (short)0, sc[tb], false, false);
            sc[tb] = __builtin_amdgcn_wmma_f32_16x16x32_bf16(
                false, qa1, false, kf1, (short)0, sc[tb], false, false);
        }

        // ---- online softmax ----
        float mnew[8], alpha[8];
#pragma unroll
        for (int r = 0; r < 8; ++r) {
            float mx = fmaxf(sc[0][r], sc[1][r]);
#pragma unroll
            for (int off = 1; off < 16; off <<= 1)
                mx = fmaxf(mx, __shfl_xor(mx, off, 32));
            mnew[r]  = fmaxf(mrow[r], mx);
            alpha[r] = __expf(mrow[r] - mnew[r]);
            mrow[r]  = mnew[r];
        }
#pragma unroll
        for (int tb = 0; tb < 2; ++tb)
#pragma unroll
            for (int r = 0; r < 8; ++r) {
                const float p = __expf(sc[tb][r] - mnew[r]);
                sc[tb][r] = p;
                ldsW[(r + 8 * hf) * 40 + tb * 16 + cn] = (bf16)p;
            }
#pragma unroll
        for (int r = 0; r < 8; ++r) {
            float rs = sc[0][r] + sc[1][r];
#pragma unroll
            for (int off = 1; off < 16; off <<= 1)
                rs += __shfl_xor(rs, off, 32);
            lrow[r] = lrow[r] * alpha[r] + rs;
#pragma unroll
            for (int j = 0; j < 4; ++j) acc[j][r] *= alpha[r];
        }

        // ---- O += P x V  (P via LDS bounce: C-layout -> A-layout) ----
        const v16bf pa = load_a_frag(ldsW, 40, 0, 0, lane);
#pragma unroll
        for (int j = 0; j < 4; ++j) {
            v16bf vb = load_bT_frag(ldsV, 40, 16 * j, 0, lane);  // B[k=t][n=d]
            acc[j] = __builtin_amdgcn_wmma_f32_16x16x32_bf16(
                false, pa, false, vb, (short)0, acc[j], false, false);
        }
        __syncthreads();
    }

    // ---- normalize and store: bf16 [B, S, H*Dh] ----
    const int b_ = bh >> 4, hd = bh & 15;
#pragma unroll
    for (int r = 0; r < 8; ++r) {
        const float inv = 1.0f / lrow[r];
        const int s = q0 + r + 8 * hf;
        const size_t base = ((size_t)(b_ * SEQ + s)) * D_MODEL + hd * DHEAD;
#pragma unroll
        for (int j = 0; j < 4; ++j)
            O[base + 16 * j + cn] = (bf16)(acc[j][r] * inv);
    }
}

// ---------------------------------------------------------------------------
// Host-side launcher
// ---------------------------------------------------------------------------
extern "C" void kernel_launch(void* const* d_in, const int* in_sizes, int n_in,
                              void* d_out, int out_size, void* d_ws, size_t ws_size,
                              hipStream_t stream) {
    const float* x  = (const float*)d_in[0];
    const float* Wq = (const float*)d_in[1];
    const float* bq = (const float*)d_in[2];
    const float* Wk = (const float*)d_in[3];
    const float* bk = (const float*)d_in[4];
    const float* Wv = (const float*)d_in[5];
    const float* bv = (const float*)d_in[6];
    const float* Wo = (const float*)d_in[7];
    const float* bo = (const float*)d_in[8];
    float* out = (float*)d_out;

    char* ws = (char*)d_ws;
    size_t off = 0;
    const size_t xbytes = (size_t)MROWS * D_MODEL * sizeof(bf16);   // 16.78 MB
    const size_t wbytes = (size_t)D_MODEL * D_MODEL * sizeof(bf16); //  2.10 MB
    bf16* xb  = (bf16*)(ws + off); off += xbytes;   // also reused as attn output
    bf16* wqb = (bf16*)(ws + off); off += wbytes;
    bf16* wkb = (bf16*)(ws + off); off += wbytes;
    bf16* wvb = (bf16*)(ws + off); off += wbytes;
    bf16* wob = (bf16*)(ws + off); off += wbytes;
    bf16* Qb  = (bf16*)(ws + off); off += xbytes;
    bf16* Kb  = (bf16*)(ws + off); off += xbytes;
    bf16* Vtb = (bf16*)(ws + off); off += xbytes;   // V transposed [B,H,Dh,S]

    // fp32 -> bf16 conversions
    const int nx = MROWS * D_MODEL;
    const int nw = D_MODEL * D_MODEL;
    cvt_f32_bf16_kernel<<<nx / 1024, 256, 0, stream>>>(x,  xb,  nx);
    cvt_f32_bf16_kernel<<<nw / 1024, 256, 0, stream>>>(Wq, wqb, nw);
    cvt_f32_bf16_kernel<<<nw / 1024, 256, 0, stream>>>(Wk, wkb, nw);
    cvt_f32_bf16_kernel<<<nw / 1024, 256, 0, stream>>>(Wv, wvb, nw);
    cvt_f32_bf16_kernel<<<nw / 1024, 256, 0, stream>>>(Wo, wob, nw);

    // Q/K/V projections (Q pre-scaled by 1/sqrt(Dh); V stored transposed)
    dim3 gg(MROWS / 256, D_MODEL / 128);
    gemm_bf16_kernel<<<gg, 256, 0, stream>>>(xb, wqb, bq, Qb,  nullptr, 0, SCALE_Q);
    gemm_bf16_kernel<<<gg, 256, 0, stream>>>(xb, wkb, bk, Kb,  nullptr, 0, 1.0f);
    gemm_bf16_kernel<<<gg, 256, 0, stream>>>(xb, wvb, bv, Vtb, nullptr, 2, 1.0f);

    // flash attention: attn output overwrites xb (x no longer needed)
    attn_kernel<<<dim3(BATCH * NHEAD, SEQ / 128), 256, 0, stream>>>(Qb, Kb, Vtb, xb);

    // final output projection -> fp32 d_out
    gemm_bf16_kernel<<<gg, 256, 0, stream>>>(xb, wob, bo, nullptr, out, 1, 1.0f);
}
